// SpatialModel_24180665877120
// MI455X (gfx1250) — matrix operand order repaired and checked by reading, not compiled, not run on hardware
//
#include <hip/hip_runtime.h>
#include <hip/hip_bf16.h>

typedef __attribute__((ext_vector_type(16))) _Float16 v16h;
typedef __attribute__((ext_vector_type(8)))  float    v8f;
typedef _Float16 half_t;

#define GAT_ALPHA 0.2f
#define NN 1024

__device__ __forceinline__ float lrelu(float x) { return fmaxf(x, GAT_ALPHA * x); }
__device__ __forceinline__ float elu1(float x)  { return x > 0.f ? x : (__expf(x) - 1.f); }

// Packed-V layout (exact WMMA B-operand register image):
//   Vpk[((bh*32 + t)*32 + lane)*16 + e]  ==  B[K = (e>>3)*16 + (lane>>4)*8 + (e&7)][col = lane&15]
// where m = t*32 + K, value = Wh[bh][m][col] for col < OCOLS, else 0.
__device__ __forceinline__ size_t vpk_index(int bh, int n, int o, int hi_from_o)
{
    const int t = n >> 5, K = n & 31, hi = (K >> 3) & 1, e = (K & 7) | ((K >> 4) << 3);
    (void)hi_from_o;
    return ((size_t)(bh * 32 + t) * 32 + (hi * 16 + o)) * 16 + e;
}

// ---------------------------------------------------------------------------
// Prep layer 1: Wh = x @ W1[h]  (F=4 -> O=2), scores, packed V (f16).
// ---------------------------------------------------------------------------
__global__ void prep1_kernel(const float* __restrict__ x,   // (32,1024,4)
                             const float* __restrict__ W1,  // (3,4,2)
                             const float* __restrict__ a1,  // (3,4,1)
                             float* __restrict__ sSrc, float* __restrict__ sDst,
                             half_t* __restrict__ Vpk)
{
    int idx = blockIdx.x * blockDim.x + threadIdx.x;   // b*N + n
    if (idx >= 32 * NN) return;
    const float4 xv = ((const float4*)x)[idx];
    const float xf[4] = {xv.x, xv.y, xv.z, xv.w};
    const int b = idx >> 10;
    const int n = idx & (NN - 1);
    #pragma unroll
    for (int h = 0; h < 3; ++h) {
        float wh0 = 0.f, wh1 = 0.f;
        #pragma unroll
        for (int f = 0; f < 4; ++f) {
            wh0 = fmaf(xf[f], W1[(h * 4 + f) * 2 + 0], wh0);
            wh1 = fmaf(xf[f], W1[(h * 4 + f) * 2 + 1], wh1);
        }
        const int bh = b * 3 + h;
        sSrc[bh * NN + n] = wh0 * a1[h * 4 + 0] + wh1 * a1[h * 4 + 1];
        sDst[bh * NN + n] = wh0 * a1[h * 4 + 2] + wh1 * a1[h * 4 + 3];
        Vpk[vpk_index(bh, n, 0, 0)] = (half_t)wh0;
        Vpk[vpk_index(bh, n, 1, 0)] = (half_t)wh1;
    }
}

// ---------------------------------------------------------------------------
// Prep layer 2: Wh = h @ W2 (F=6 -> O=4), single head.
// ---------------------------------------------------------------------------
__global__ void prep2_kernel(const float* __restrict__ hin, // (32,1024,6)
                             const float* __restrict__ W2,  // (1,6,4)
                             const float* __restrict__ a2,  // (1,8,1)
                             float* __restrict__ sSrc, float* __restrict__ sDst,
                             half_t* __restrict__ Vpk)
{
    int idx = blockIdx.x * blockDim.x + threadIdx.x;   // b*N + n == bh*N + n
    if (idx >= 32 * NN) return;
    float hf[6];
    #pragma unroll
    for (int f = 0; f < 6; ++f) hf[f] = hin[idx * 6 + f];
    float wh[4] = {0.f, 0.f, 0.f, 0.f};
    #pragma unroll
    for (int f = 0; f < 6; ++f)
        #pragma unroll
        for (int o = 0; o < 4; ++o)
            wh[o] = fmaf(hf[f], W2[f * 4 + o], wh[o]);
    float ss = 0.f, sd = 0.f;
    #pragma unroll
    for (int o = 0; o < 4; ++o) {
        ss = fmaf(wh[o], a2[o],     ss);
        sd = fmaf(wh[o], a2[4 + o], sd);
    }
    sSrc[idx] = ss;
    sDst[idx] = sd;
    const int bh = idx >> 10, n = idx & (NN - 1);
    #pragma unroll
    for (int o = 0; o < 4; ++o) Vpk[vpk_index(bh, n, o, 0)] = (half_t)wh[o];
}

// ---------------------------------------------------------------------------
// Per-(b,h) max over s_dst (exact row-max of e since lrelu is monotone).
// ---------------------------------------------------------------------------
__global__ void maxreduce_kernel(const float* __restrict__ s, float* __restrict__ out)
{
    __shared__ float sm[256];
    const int bh = blockIdx.x, t = threadIdx.x;
    const float* p = s + bh * NN;
    float m = fmaxf(fmaxf(p[t], p[t + 256]), fmaxf(p[t + 512], p[t + 768]));
    sm[t] = m;
    __syncthreads();
    for (int w = 128; w > 0; w >>= 1) {
        if (t < w) sm[t] = fmaxf(sm[t], sm[t + w]);
        __syncthreads();
    }
    if (t == 0) out[bh] = sm[0];
}

// ---------------------------------------------------------------------------
// Fused softmax-attention: one wave per 16-row n-tile; K loop over m in 32s.
//   P[n,m] = exp(lrelu(sS[n]+sD[m]) - shift[n])   (f16, <= 1.0)
//   numer  = P @ V  via v_wmma_f32_16x16x32_f16   (f32 accumulate)
//   denom  = sum_m P (exact f32 VALU; lane pair combined by shfl_xor 16)
// A layout (16b A 16x32): lane<16 -> M=lane, K(e) = (e<8 ? e : e+8);  lane>=16: +8
// B layout (16b B 32x16): N=lane&15, K = (e<8 ? e : e+8) + (lane>=16?8:0)  (pre-packed in Vpk)
// D layout: lane column N=lane&15, rows M = j + (lane>=16?8:0), j=0..7
// ---------------------------------------------------------------------------
template <int OCOLS, int HEADS, int OSTRIDE>
__global__ void __launch_bounds__(256) attn_kernel(
    const float* __restrict__ sSrc, const float* __restrict__ sDst,
    const half_t* __restrict__ Vpk, const float* __restrict__ maxDst,
    float* __restrict__ out)
{
    const int lane = threadIdx.x & 31;
    const int wave = threadIdx.x >> 5;
    const int tile = blockIdx.x * 8 + wave;   // 8 waves per block
    const int bh = tile >> 6;                 // 64 n-tiles per (b,h)
    const int nT = tile & 63;
    const int b = bh / HEADS;                 // compile-time constant division
    const int head = bh - b * HEADS;

    const float*  sS = sSrc + bh * NN;
    const float*  sD = sDst + bh * NN;
    const half_t* vb = Vpk + ((size_t)bh * 32 * 32 + lane) * 16; // per-lane B column

    const int rowA = lane & 15;               // A-row / D-column index
    const int hi   = lane >> 4;               // lane-half selector
    const int kOff = hi * 8;
    const float ss = sS[nT * 16 + rowA];
    const float shift = lrelu(ss + maxDst[bh]);

    v8f acc = {};
    float denom = 0.f;

    for (int t = 0; t < 32; ++t) {
        const int mBase = t * 32;
        // ---- s_dst slices for this lane's K set: two runs of 8 floats ----
        const float4 d0 = *(const float4*)(sD + mBase + kOff);
        const float4 d1 = *(const float4*)(sD + mBase + kOff + 4);
        const float4 d2 = *(const float4*)(sD + mBase + kOff + 16);
        const float4 d3 = *(const float4*)(sD + mBase + kOff + 20);
        const float dv[16] = {d0.x, d0.y, d0.z, d0.w, d1.x, d1.y, d1.z, d1.w,
                              d2.x, d2.y, d2.z, d2.w, d3.x, d3.y, d3.z, d3.w};
        // ---- A operand: probability tile (f16) + f32 denominator ----
        v16h aPack;
        #pragma unroll
        for (int e = 0; e < 16; ++e) {
            const float p = __expf(lrelu(ss + dv[e]) - shift);
            denom += p;
            aPack[e] = (half_t)p;
        }
        // ---- B operand: single contiguous 32B load (pre-packed layout) ----
        const v16h bPack = *(const v16h*)(vb + (size_t)t * 32 * 16);
        // ---- D += A x B ----
        acc = __builtin_amdgcn_wmma_f32_16x16x32_f16(
            /*neg_a=*/false, aPack, /*neg_b=*/false, bPack,
            /*c_mod=*/(short)0, acc, /*reuse_a=*/false, /*reuse_b=*/false);
    }

    // Row denominators: lanes L and L^16 hold disjoint K-halves of row L&15.
    denom += __shfl_xor(denom, 16, 32);

    // Normalize + ELU + scatter. Lane owns column `col`, rows j+hi*8.
    const int col = lane & 15;
    #pragma unroll
    for (int j = 0; j < 8; ++j) {
        const int r = j + hi * 8;
        const float dnm = __shfl(denom, r, 32);   // lane r holds denom of row r
        const float v = acc[j] / dnm;
        if (col < OCOLS) {
            const int nOut = nT * 16 + r;
            out[(size_t)(b * NN + nOut) * OSTRIDE + head * OCOLS + col] = elu1(v);
        }
    }
}

// ---------------------------------------------------------------------------
extern "C" void kernel_launch(void* const* d_in, const int* in_sizes, int n_in,
                              void* d_out, int out_size, void* d_ws, size_t ws_size,
                              hipStream_t stream)
{
    const float* x  = (const float*)d_in[0];
    const float* W1 = (const float*)d_in[1];
    const float* a1 = (const float*)d_in[2];
    const float* W2 = (const float*)d_in[3];
    const float* a2 = (const float*)d_in[4];

    // Carve workspace (256B aligned slices)
    char* p = (char*)d_ws;
    auto carve = [&](size_t bytes) -> void* {
        void* r = (void*)p;
        p += (bytes + 255) & ~(size_t)255;
        return r;
    };
    const size_t vpk1Bytes = (size_t)96 * 32 * 32 * 16 * sizeof(half_t); // 3 MB
    const size_t vpk2Bytes = (size_t)32 * 32 * 32 * 16 * sizeof(half_t); // 1 MB
    float*  sSrc1 = (float*)carve(96 * NN * sizeof(float));
    float*  sDst1 = (float*)carve(96 * NN * sizeof(float));
    half_t* Vpk1  = (half_t*)carve(vpk1Bytes);
    float*  maxD1 = (float*)carve(96 * sizeof(float));
    float*  h1    = (float*)carve(32 * NN * 6 * sizeof(float));
    float*  sSrc2 = (float*)carve(32 * NN * sizeof(float));
    float*  sDst2 = (float*)carve(32 * NN * sizeof(float));
    half_t* Vpk2  = (half_t*)carve(vpk2Bytes);
    float*  maxD2 = (float*)carve(32 * sizeof(float));
    (void)ws_size; (void)n_in; (void)in_sizes; (void)out_size;

    // Zero packed-V (columns >= OCOLS must be 0); capture-legal, every call.
    hipMemsetAsync(Vpk1, 0, vpk1Bytes, stream);
    hipMemsetAsync(Vpk2, 0, vpk2Bytes, stream);

    // Layer 1
    prep1_kernel<<<128, 256, 0, stream>>>(x, W1, a1, sSrc1, sDst1, Vpk1);
    maxreduce_kernel<<<96, 256, 0, stream>>>(sDst1, maxD1);
    attn_kernel<2, 3, 6><<<768, 256, 0, stream>>>(sSrc1, sDst1, Vpk1, maxD1, h1);

    // Layer 2
    prep2_kernel<<<128, 256, 0, stream>>>(h1, W2, a2, sSrc2, sDst2, Vpk2);
    maxreduce_kernel<<<32, 256, 0, stream>>>(sDst2, maxD2);
    attn_kernel<4, 1, 4><<<256, 256, 0, stream>>>(sSrc2, sDst2, Vpk2, maxD2, (float*)d_out);
}
